// WindowMHSA_59201829208371
// MI455X (gfx1250) — compile-verified, matrix operand order checked by reading, not run on hardware
//
#include <hip/hip_runtime.h>

typedef __attribute__((ext_vector_type(16))) _Float16 v16h;
typedef __attribute__((ext_vector_type(8)))  _Float16 v8h;
typedef __attribute__((ext_vector_type(8)))  float    v8f;

#define WSZ   7
#define NPIX  49
#define NPAD  64
#define CDIM  128
#define NHEAD 4
#define HDIM  32
#define IMH   224
#define IMW   224

#define WQKV_ELTS  (384 * 128)          // 49152
#define WPROJ_ELTS (128 * 128)          // 16384
#define WTOT_ELTS  (WQKV_ELTS + WPROJ_ELTS)

// LDS layout (bytes); all offsets 256-aligned, all leading dims keep 16B alignment
#define OFF_XT   0        // sXt : [64][136] f16  (pixel, channel), padded ld
#define OFF_Q    17408    // sQ  : [4][64][32] f16 (head, pixel, d)  -- scaled
#define OFF_K    33792    // sK  : [4][64][32] f16 (head, pixel, d)
#define OFF_VT   50176    // sVt : [4][32][64] f16 (head, d, pixel)
#define OFF_S    66560    // sS  : [4][64][64] f32 (head, qpix, kpix)
#define OFF_P    132096   // sP  : [4][64][64] f16
#define OFF_AT   164864   // sAt : [64][136] f16 (pixel, channel), padded ld
#define LDS_BYTES 182272

// Fragment gather for 16-bit WMMA operands (CDNA5 ISA 7.12.2):
// lanes 0-15 hold K 0..7 / 16..23, lanes 16-31 hold K 8..15 / 24..31.
// Works for A ([M][K] storage) and for B stored K-contiguous ([N][K] storage).
// Two 16-byte loads per lane -> ds_load_b128 (LDS) or global_load_b128 (global).
__device__ __forceinline__ v16h frag16(const _Float16* src, int ldk,
                                       int row0, int k0, int lane) {
    int r  = row0 + (lane & 15);
    int hs = (lane >> 4) & 1;
    const v8h lo = *(const v8h*)(src + (size_t)r * ldk + k0 + hs * 8);
    const v8h hi = *(const v8h*)(src + (size_t)r * ldk + k0 + 16 + hs * 8);
    return __builtin_shufflevector(lo, hi, 0,1,2,3,4,5,6,7,8,9,10,11,12,13,14,15);
}

__device__ __forceinline__ v8f wmma_f16(v16h a, v16h b, v8f c) {
    return __builtin_amdgcn_wmma_f32_16x16x32_f16(false, a, false, b,
                                                  (short)0, c, false, false);
}

// One-time fp32 -> f16 weight conversion into workspace (weights are tiny and
// L2-resident; converting once removes 8192x redundant v_cvt work from the
// main kernel's inner GEMM loops and halves weight fetch bytes).
__global__ __launch_bounds__(256)
void convert_weights(const float* __restrict__ wqkv,
                     const float* __restrict__ wproj,
                     _Float16* __restrict__ w16) {
    int i = blockIdx.x * 256 + threadIdx.x;
    if (i < WQKV_ELTS)      w16[i] = (_Float16)wqkv[i];
    else if (i < WTOT_ELTS) w16[i] = (_Float16)wproj[i - WQKV_ELTS];
}

__global__ __launch_bounds__(256, 1)
void window_mhsa_fused(const float* __restrict__ x,
                       const _Float16* __restrict__ wqkv16,
                       const _Float16* __restrict__ wproj16,
                       float* __restrict__ out) {
    extern __shared__ char smem[];
    _Float16* sXt = (_Float16*)(smem + OFF_XT);
    _Float16* sQ  = (_Float16*)(smem + OFF_Q);
    _Float16* sK  = (_Float16*)(smem + OFF_K);
    _Float16* sVt = (_Float16*)(smem + OFF_VT);
    float*    sS  = (float*)   (smem + OFF_S);
    _Float16* sP  = (_Float16*)(smem + OFF_P);
    _Float16* sAt = (_Float16*)(smem + OFF_AT);

    const int tid  = threadIdx.x;
    const int lane = tid & 31;
    const int wave = tid >> 5;

    const int blk = blockIdx.x;           // 8192 windows
    const int b   = blk >> 10;
    const int wh  = (blk >> 5) & 31;
    const int ww  = blk & 31;
    const int h0  = wh * WSZ, w0 = ww * WSZ;

    // CDNA5 prefetch of the f16 weight rows this wave will consume
    if (lane == 0) {
        __builtin_prefetch(wqkv16  + (size_t)wave * 48 * CDIM, 0, 0);
        __builtin_prefetch(wproj16 + (size_t)wave * 16 * CDIM, 0, 0);
    }

    // ---- Phase 0: window load, f32 -> f16, transposed to [pix][chan] ----
    for (int i = tid; i < NPAD * CDIM; i += 256) {
        int p = i & 63;              // pixel fastest: coalesced 7-wide row reads
        int c = i >> 6;
        float v = 0.f;
        if (p < NPIX) {
            int py = p / WSZ, px = p % WSZ;
            v = x[(((size_t)b * CDIM + c) * IMH + h0 + py) * IMW + w0 + px];
        }
        sXt[p * 136 + c] = (_Float16)v;
    }
    __syncthreads();

    // ---- Phase 1: QKV = w_qkv(384x128) @ X(128x64). wave -> 48 output rows ----
    const float qscale = 0.1767766952966369f;   // 1/sqrt(32)
    #pragma unroll
    for (int mt = 0; mt < 3; ++mt) {
        const int o0 = wave * 48 + mt * 16;
        v8f a0 = {}, a1 = {}, a2 = {}, a3 = {};
        #pragma unroll
        for (int ks = 0; ks < 4; ++ks) {
            const int k0 = ks * 32;
            v16h af = frag16(wqkv16, CDIM, o0, k0, lane);   // global_load_b128 x2
            a0 = wmma_f16(af, frag16(sXt, 136,  0, k0, lane), a0);
            a1 = wmma_f16(af, frag16(sXt, 136, 16, k0, lane), a1);
            a2 = wmma_f16(af, frag16(sXt, 136, 32, k0, lane), a2);
            a3 = wmma_f16(af, frag16(sXt, 136, 48, k0, lane), a3);
        }
        v8f accs[4] = {a0, a1, a2, a3};
        #pragma unroll
        for (int nt = 0; nt < 4; ++nt) {
            #pragma unroll
            for (int r = 0; r < 8; ++r) {
                const int o = o0 + r + 8 * (lane >> 4);   // qkv output row
                const int n = nt * 16 + (lane & 15);      // pixel
                const int c = o & 127, which = o >> 7;
                const int head = c >> 5, d = c & 31;
                const float val = accs[nt][r];
                if (which == 0)      sQ [(head * 64 + n) * 32 + d] = (_Float16)(val * qscale);
                else if (which == 1) sK [(head * 64 + n) * 32 + d] = (_Float16)val;
                else                 sVt[(head * 32 + d) * 64 + n] = (_Float16)val;
            }
        }
    }
    __syncthreads();

    // ---- Phase 2: scores = Q @ K^T per head (K-dim = 32) ----
    {
        const int head = wave >> 1, half = wave & 1;
        const _Float16* q = sQ + head * 64 * 32;
        const _Float16* k = sK + head * 64 * 32;
        float* s = sS + head * 64 * 64;
        #pragma unroll
        for (int mt = 0; mt < 2; ++mt) {
            const int m0 = half * 32 + mt * 16;
            v16h af = frag16(q, 32, m0, 0, lane);
            #pragma unroll
            for (int nt = 0; nt < 4; ++nt) {
                v8f acc = {};
                acc = wmma_f16(af, frag16(k, 32, nt * 16, 0, lane), acc);
                #pragma unroll
                for (int r = 0; r < 8; ++r) {
                    const int m = m0 + r + 8 * (lane >> 4);
                    const int n = nt * 16 + (lane & 15);
                    s[m * 64 + n] = acc[r];
                }
            }
        }
    }
    __syncthreads();

    // ---- Phase 3: softmax over heads (reference: softmax(axis=-1) on 'bnmh') ----
    // Padded key pixels have V == 0, so no masking is needed.
    for (int pos = tid; pos < 64 * 64; pos += 256) {
        float s0 = sS[0 * 4096 + pos], s1 = sS[1 * 4096 + pos];
        float s2 = sS[2 * 4096 + pos], s3 = sS[3 * 4096 + pos];
        float mx = fmaxf(fmaxf(s0, s1), fmaxf(s2, s3));
        float e0 = __expf(s0 - mx), e1 = __expf(s1 - mx);
        float e2 = __expf(s2 - mx), e3 = __expf(s3 - mx);
        float inv = 1.f / (e0 + e1 + e2 + e3);
        sP[0 * 4096 + pos] = (_Float16)(e0 * inv);
        sP[1 * 4096 + pos] = (_Float16)(e1 * inv);
        sP[2 * 4096 + pos] = (_Float16)(e2 * inv);
        sP[3 * 4096 + pos] = (_Float16)(e3 * inv);
    }
    __syncthreads();

    // ---- Phase 4: out_head = P(64x64) @ V(64x32) per head ----
    {
        const int head = wave >> 1, half = wave & 1;
        const _Float16* p = sP  + head * 4096;
        const _Float16* v = sVt + head * 32 * 64;   // [d][pix] = B^T storage
        #pragma unroll
        for (int mt = 0; mt < 2; ++mt) {
            const int m0 = half * 32 + mt * 16;
            v8f a0 = {}, a1 = {};
            #pragma unroll
            for (int ks = 0; ks < 2; ++ks) {
                const int k0 = ks * 32;
                v16h af = frag16(p, 64, m0, k0, lane);
                a0 = wmma_f16(af, frag16(v, 64,  0, k0, lane), a0);
                a1 = wmma_f16(af, frag16(v, 64, 16, k0, lane), a1);
            }
            #pragma unroll
            for (int r = 0; r < 8; ++r) {
                const int m  = m0 + r + 8 * (lane >> 4);
                const int dl = lane & 15;
                sAt[m * 136 + head * 32 + dl]      = (_Float16)a0[r];
                sAt[m * 136 + head * 32 + 16 + dl] = (_Float16)a1[r];
            }
        }
    }
    __syncthreads();

    // ---- Phase 5: out = w_proj(128x128) @ attn(128x64); wave -> 16 channels ----
    {
        const int o0 = wave * 16;
        v8f a0 = {}, a1 = {}, a2 = {}, a3 = {};
        #pragma unroll
        for (int ks = 0; ks < 4; ++ks) {
            const int k0 = ks * 32;
            v16h af = frag16(wproj16, CDIM, o0, k0, lane);  // global_load_b128 x2
            a0 = wmma_f16(af, frag16(sAt, 136,  0, k0, lane), a0);
            a1 = wmma_f16(af, frag16(sAt, 136, 16, k0, lane), a1);
            a2 = wmma_f16(af, frag16(sAt, 136, 32, k0, lane), a2);
            a3 = wmma_f16(af, frag16(sAt, 136, 48, k0, lane), a3);
        }
        v8f accs[4] = {a0, a1, a2, a3};
        #pragma unroll
        for (int nt = 0; nt < 4; ++nt) {
            #pragma unroll
            for (int r = 0; r < 8; ++r) {
                const int oc  = o0 + r + 8 * (lane >> 4);
                const int pix = nt * 16 + (lane & 15);
                if (pix < NPIX) {
                    const int py = pix / WSZ, px = pix % WSZ;
                    out[(((size_t)b * CDIM + oc) * IMH + h0 + py) * IMW + w0 + px] = accs[nt][r];
                }
            }
        }
    }
}

extern "C" void kernel_launch(void* const* d_in, const int* in_sizes, int n_in,
                              void* d_out, int out_size, void* d_ws, size_t ws_size,
                              hipStream_t stream) {
    (void)in_sizes; (void)n_in; (void)out_size; (void)ws_size;
    const float* x     = (const float*)d_in[0];
    const float* wqkv  = (const float*)d_in[1];
    const float* wproj = (const float*)d_in[2];
    float* out = (float*)d_out;

    // f16 weight cache in workspace: 384*128 + 128*128 halves = 128 KB
    _Float16* w16 = (_Float16*)d_ws;

    convert_weights<<<dim3((WTOT_ELTS + 255) / 256), dim3(256), 0, stream>>>(wqkv, wproj, w16);

    const int nwin = 8 * 32 * 32;   // B * nWh * nWw = 8192 windows
    window_mhsa_fused<<<dim3(nwin), dim3(256), LDS_BYTES, stream>>>(
        x, w16, w16 + WQKV_ELTS, out);
}
